// DifferentialEquation_70935679861075
// MI455X (gfx1250) — matrix operand order repaired and checked by reading, not compiled, not run on hardware
//
#include <hip/hip_runtime.h>
#include <math.h>

// ---------------------------------------------------------------------------
// Neural CDE solver for MI455X (gfx1250), wave32 + V_WMMA_F32_16X16X32_F16.
// 16 workgroups x 512 threads; each WG owns 16 batch rows and runs all 2048
// steps with weights resident in LDS as f16 (f32 accumulate in WMMA).
// Activations use v_exp_f32 + v_rcp_f32; k-reduction uses dual independent
// WMMA accumulators to avoid the WMMA->WMMA RAW hazard NOPs.
// ---------------------------------------------------------------------------

typedef __attribute__((ext_vector_type(16))) _Float16 v16h;
typedef __attribute__((ext_vector_type(8)))  _Float16 v8h;
typedef __attribute__((ext_vector_type(8)))  float    v8f;

union Vu { v16h v; v8h h[2]; };

#define THREADS 512
#define NWAVES  16
#define MT      16          // batch rows per workgroup
#define NBATCH  256
#define NSTEPS  2048
#define TSTEPS  (NSTEPS + 1)
#define HID     64
#define WID     128
#define CSZ     8

// padded row strides (halves) -> conflict-free 16-lane b128 fragment loads
#define S64  72
#define S128 136

// ---- LDS layout: f16 region (offsets in halves) ----
#define OFF_VFW1   0
#define OFF_CVFW1  (OFF_VFW1  + 128*S64)
#define OFF_VFW2   (OFF_CVFW1 + 128*S64)
#define OFF_CVFW2  (OFF_VFW2  + 128*S128)
#define OFF_VFW3   (OFF_CVFW2 + 128*S128)
#define OFF_CVFW3  (OFF_VFW3  + 64*S128)
#define OFF_HIN    (OFF_CVFW3 + 512*S128)
#define OFF_VFH1   (OFF_HIN   + MT*S64)
#define OFF_CVFH1  (OFF_VFH1  + MT*S128)
#define OFF_VFH2   (OFF_CVFH1 + MT*S128)
#define OFF_CVFH2  (OFF_VFH2  + MT*S128)
#define OFF_FBUF   (OFF_CVFH2 + MT*S128)
#define OFF_GBUF   (OFF_FBUF  + MT*HID)
#define HALVES_END (OFF_GBUF  + MT*512)

// ---- LDS layout: f32 region (offsets in floats), begins at 2*HALVES_END ----
#define F_VFB1  0
#define F_CVFB1 (F_VFB1  + 128)
#define F_VFB2  (F_CVFB1 + 128)
#define F_CVFB2 (F_VFB2  + 128)
#define F_VFB3  (F_CVFB2 + 128)
#define F_CVFB3 (F_VFB3  + 64)
#define F_TCV   (F_CVFB3 + 512)
#define F_TCC   (F_TCV   + 128)
#define F_BEFF  (F_TCC   + 128)
#define F_RW    (F_BEFF  + 256)
#define F_RB    (F_RW    + 64)
#define F_YBUF  (F_RB    + 16)
#define F_YHAT  (F_YBUF  + MT*HID)
#define F_DX    (F_YHAT  + MT*HID)
#define F_P0    (F_DX    + MT*CSZ)
#define F_END   (F_P0    + MT*HID)

#define SMEM_BYTES (2*HALVES_END + 4*F_END)
static_assert(SMEM_BYTES <= 320 * 1024, "LDS budget exceeded");

// ---------------------------------------------------------------------------
// Fast activations: v_exp_f32 + v_rcp_f32, no IEEE division.
// tanh(x) = 2*sigmoid(2x) - 1 saturates correctly (exp->inf => rcp->0).
// ---------------------------------------------------------------------------
__device__ __forceinline__ float fast_sigmoid(float x) {
#if defined(__AMDGCN__)
  return __builtin_amdgcn_rcpf(1.0f + __expf(-x));
#else
  return 1.0f / (1.0f + expf(-x));
#endif
}
__device__ __forceinline__ float fast_tanh(float x) {
  return 2.0f * fast_sigmoid(2.0f * x) - 1.0f;
}
__device__ __forceinline__ float lipswish(float x) {
  return 0.909f * x * fast_sigmoid(x);
}

// ---------------------------------------------------------------------------
// WMMA fragment helpers. Layouts per CDNA5 ISA 7.12.2:
//  A (16x32 f16): lane<16 row=lane holds K [k0..k0+7],[k0+16..k0+23];
//                 lane>=16 row=lane-16 holds K [k0+8..k0+15],[k0+24..k0+31].
//  B (32x16 f16): lane n<16 holds K [k0..k0+7],[k0+16..k0+23] of column n
//                 (row-major W[N][K]); lane>=16 holds K [k0+8..k0+15],[k0+24..k0+31].
//  D (16x16 f32): vgpr r -> row r (lanes 0-15, col=lane) / row r+8 (lanes>=16).
// Dual accumulators (even/odd k-chunk) keep adjacent WMMAs independent so the
// compiler need not insert WMMA->WMMA hazard NOPs between them.
// ---------------------------------------------------------------------------
template <int KCHUNKS>
__device__ __forceinline__ v8f mm_tile(const _Float16* __restrict__ A, int sA,
                                       const _Float16* __restrict__ W, int sW,
                                       int n0, int lane) {
  v8f acc0 = {0.f, 0.f, 0.f, 0.f, 0.f, 0.f, 0.f, 0.f};
  v8f acc1 = acc0;
  const int row = lane & 15;
  const int hi  = (lane & 16) ? 1 : 0;
#pragma unroll
  for (int kc = 0; kc < KCHUNKS; ++kc) {
    const int k0 = kc * 32;
    Vu a, b;
    const _Float16* pa = A + row * sA + k0 + hi * 8;
    a.h[0] = *(const v8h*)(pa);
    a.h[1] = *(const v8h*)(pa + 16);
    const _Float16* pb = W + (n0 + row) * sW + k0 + hi * 8;
    b.h[0] = *(const v8h*)(pb);
    b.h[1] = *(const v8h*)(pb + 16);
#if defined(__AMDGCN__)
    if (kc & 1)
      acc1 = __builtin_amdgcn_wmma_f32_16x16x32_f16(false, a.v, false, b.v,
                                                    (short)0, acc1, false, false);
    else
      acc0 = __builtin_amdgcn_wmma_f32_16x16x32_f16(false, a.v, false, b.v,
                                                    (short)0, acc0, false, false);
#else
    (void)a; (void)b;
#endif
  }
  return (KCHUNKS > 1) ? (acc0 + acc1) : acc0;
}

// act: 0 = lipswish (0.909 * silu), 1 = tanh
__device__ __forceinline__ void store_tile(v8f acc, _Float16* __restrict__ dst,
                                           int ld, int n0, int lane,
                                           const float* __restrict__ bias, int act) {
  const int col = lane & 15;
  const int rb8 = (lane & 16) ? 8 : 0;
  const float bn = bias[n0 + col];
#pragma unroll
  for (int r = 0; r < 8; ++r) {
    float x = acc[r] + bn;
    x = (act == 0) ? lipswish(x) : fast_tanh(x);
    dst[(rb8 + r) * ld + n0 + col] = (_Float16)x;
  }
}

// Evaluate vf and cvf at time t from activations already in hin (f16, 16x64).
// Results: fbuf (16x64 f16, tanh), gbuf (16x512 f16, tanh).
__device__ __forceinline__ void eval_nets(char* smemraw, float t,
                                          int tid, int lane, int wave) {
  _Float16* H  = (_Float16*)smemraw;
  float*    Fp = (float*)(smemraw + 2 * HALVES_END);
  _Float16 *vfW1 = H + OFF_VFW1, *cvfW1 = H + OFF_CVFW1;
  _Float16 *vfW2 = H + OFF_VFW2, *cvfW2 = H + OFF_CVFW2;
  _Float16 *vfW3 = H + OFF_VFW3, *cvfW3 = H + OFF_CVFW3;
  _Float16 *hin  = H + OFF_HIN;
  _Float16 *vfh1 = H + OFF_VFH1, *cvfh1 = H + OFF_CVFH1;
  _Float16 *vfh2 = H + OFF_VFH2, *cvfh2 = H + OFF_CVFH2;
  _Float16 *fbuf = H + OFF_FBUF, *gbuf  = H + OFF_GBUF;
  float *beff = Fp + F_BEFF;

  // Layer-1 effective bias: fold the batch-uniform t column in.
  if (tid < 256) {
    float b = (tid < 128) ? (Fp[F_VFB1 + tid]        + Fp[F_TCV + tid] * t)
                          : (Fp[F_CVFB1 + tid - 128] + Fp[F_TCC + tid - 128] * t);
    beff[tid] = b;
  }
  __syncthreads();

  // L1: combined 256-wide (vf || cvf), K=64. One 16x16 tile per wave.
  {
    const bool isV = wave < 8;
    const int  n0  = (wave & 7) * 16;
    v8f acc = mm_tile<2>(hin, S64, isV ? vfW1 : cvfW1, S64, n0, lane);
    store_tile(acc, isV ? vfh1 : cvfh1, S128, n0, lane,
               beff + (isV ? 0 : 128), 0);
  }
  __syncthreads();

  // L2: K=128. waves 0-7 -> vf tiles, waves 8-15 -> cvf tiles.
  {
    const bool isV = wave < 8;
    const int  n0  = (wave & 7) * 16;
    v8f acc = mm_tile<4>(isV ? vfh1 : cvfh1, S128,
                         isV ? vfW2 : cvfW2, S128, n0, lane);
    store_tile(acc, isV ? vfh2 : cvfh2, S128, n0, lane,
               Fp + (isV ? F_VFB2 : F_CVFB2), 0);
  }
  __syncthreads();

  // L3: 4 vf tiles (N=64) + 32 cvf tiles (N=512), round-robin over 16 waves.
  for (int tt = wave; tt < 36; tt += NWAVES) {
    if (tt < 4) {
      v8f acc = mm_tile<4>(vfh2, S128, vfW3, S128, tt * 16, lane);
      store_tile(acc, fbuf, HID, tt * 16, lane, Fp + F_VFB3, 1);
    } else {
      const int n0 = (tt - 4) * 16;
      v8f acc = mm_tile<4>(cvfh2, S128, cvfW3, S128, n0, lane);
      store_tile(acc, gbuf, 512, n0, lane, Fp + F_CVFB3, 1);
    }
  }
  __syncthreads();
}

__device__ __forceinline__ void cvt_w(const float* __restrict__ g, _Float16* l,
                                      int N, int K, int ld, int gld, int gc0,
                                      int tid) {
  for (int i = tid; i < N * K; i += THREADS) {
    const int n = i / K, k = i - n * K;
    l[n * ld + k] = (_Float16)g[n * gld + gc0 + k];
  }
}

__global__ __launch_bounds__(THREADS)
void ncde_kernel(const float* __restrict__ ts, const float* __restrict__ xinit,
                 const float* __restrict__ control,
                 const float* __restrict__ vW1, const float* __restrict__ vb1,
                 const float* __restrict__ vW2, const float* __restrict__ vb2,
                 const float* __restrict__ vW3, const float* __restrict__ vb3,
                 const float* __restrict__ cW1, const float* __restrict__ cb1,
                 const float* __restrict__ cW2, const float* __restrict__ cb2,
                 const float* __restrict__ cW3, const float* __restrict__ cb3,
                 const float* __restrict__ iW1, const float* __restrict__ ib1,
                 const float* __restrict__ iW2, const float* __restrict__ ib2,
                 const float* __restrict__ iW3, const float* __restrict__ ib3,
                 const float* __restrict__ rW,  const float* __restrict__ rb,
                 float* __restrict__ out) {
  extern __shared__ __align__(16) char smemraw[];
  _Float16* H  = (_Float16*)smemraw;
  float*    Fp = (float*)(smemraw + 2 * HALVES_END);

  const int tid  = threadIdx.x;
  const int lane = tid & 31;
  const int wave = tid >> 5;
  const int row0 = blockIdx.x * MT;

  _Float16 *hin = H + OFF_HIN, *fbuf = H + OFF_FBUF, *gbuf = H + OFF_GBUF;
  float *ybuf = Fp + F_YBUF, *yhbuf = Fp + F_YHAT;
  float *dxs = Fp + F_DX, *p0buf = Fp + F_P0;

  // ---- one-time weight preload: f32 global -> f16 LDS (padded strides) ----
  cvt_w(vW1, H + OFF_VFW1, 128, 64, S64, 65, 1, tid);   // skip t column
  cvt_w(cW1, H + OFF_CVFW1, 128, 64, S64, 65, 1, tid);
  cvt_w(vW2, H + OFF_VFW2, 128, 128, S128, 128, 0, tid);
  cvt_w(cW2, H + OFF_CVFW2, 128, 128, S128, 128, 0, tid);
  cvt_w(vW3, H + OFF_VFW3, 64, 128, S128, 128, 0, tid);
  cvt_w(cW3, H + OFF_CVFW3, 512, 128, S128, 128, 0, tid);
  for (int i = tid; i < 128; i += THREADS) {
    Fp[F_VFB1 + i] = vb1[i];  Fp[F_CVFB1 + i] = cb1[i];
    Fp[F_VFB2 + i] = vb2[i];  Fp[F_CVFB2 + i] = cb2[i];
    Fp[F_TCV + i] = vW1[i * 65];  Fp[F_TCC + i] = cW1[i * 65];
  }
  for (int i = tid; i < 64; i += THREADS) {
    Fp[F_VFB3 + i] = vb3[i];  Fp[F_RW + i] = rW[i];
  }
  for (int i = tid; i < 512; i += THREADS) Fp[F_CVFB3 + i] = cb3[i];
  if (tid == 0) Fp[F_RB] = rb[0];
  __syncthreads();

  // ---- initial network y0 = MLP(init), relu, 8->128->128->64 (runs once) ---
  {
    float* tA = (float*)gbuf;          // reuse gbuf region as 2x 16x128 f32
    float* tB = tA + MT * WID;
    for (int j = tid; j < MT * WID; j += THREADS) {
      const int m = j >> 7, n = j & 127;
      const float* wr = iW1 + n * 8;
      const float* xr = xinit + (size_t)(row0 + m) * 8;
      float a = ib1[n];
      for (int k = 0; k < 8; ++k) a += wr[k] * xr[k];
      tA[j] = fmaxf(a, 0.f);
    }
    __syncthreads();
    for (int j = tid; j < MT * WID; j += THREADS) {
      const int m = j >> 7, n = j & 127;
      const float* wr = iW2 + n * 128;
      const float* xr = tA + m * 128;
      float a = ib2[n];
      for (int k = 0; k < 128; ++k) a += wr[k] * xr[k];
      tB[j] = fmaxf(a, 0.f);
    }
    __syncthreads();
    for (int j = tid; j < MT * HID; j += THREADS) {
      const int m = j >> 6, h = j & 63;
      const float* wr = iW3 + h * 128;
      const float* xr = tB + m * 128;
      float a = ib3[h];
      for (int k = 0; k < 128; ++k) a += wr[k] * xr[k];
      ybuf[j] = a;  yhbuf[j] = a;
      hin[m * S64 + h] = (_Float16)a;
    }
    __syncthreads();
    if (tid < MT) {  // readout of y0 -> out[0] (bank-rotated traversal)
      float a = Fp[F_RB];
      for (int h = 0; h < HID; ++h) {
        const int hh = (h + tid * 4) & 63;
        a += ybuf[tid * HID + hh] * Fp[F_RW + hh];
      }
      out[row0 + tid] = a;
    }
  }

  // ---- f0, g0 at t = ts[0] ----
  eval_nets(smemraw, ts[0], tid, lane, wave);

  // ---- main sequential scan ----
  for (int i = 0; i < NSTEPS; ++i) {
    const float t1 = ts[i + 1];
    const float dt = t1 - ts[i];
    if (tid < MT * CSZ) {  // dx = control[:, i+1] - control[:, i]
      const int m = tid >> 3, cc = tid & 7;
      const size_t base = ((size_t)(row0 + m) * TSTEPS + i) * CSZ + cc;
      dxs[tid] = control[base + CSZ] - control[base];
    }
    __syncthreads();

    // p0 = f*dt + g.dx ; yhat1 = 2y - yhat + p0 ; hin <- f16(yhat1)
#pragma unroll
    for (int rep = 0; rep < 2; ++rep) {
      const int j = tid + rep * THREADS;
      const int m = j >> 6, h = j & 63;
      float a = (float)fbuf[j] * dt;
      const _Float16* gr = gbuf + (m * 512 + h * 8);
      const float* dxr = dxs + m * 8;
#pragma unroll
      for (int cc = 0; cc < 8; ++cc) a += (float)gr[cc] * dxr[cc];
      p0buf[j] = a;
      const float yh1 = 2.f * ybuf[j] - yhbuf[j] + a;
      yhbuf[j] = yh1;
      hin[m * S64 + h] = (_Float16)yh1;
    }

    eval_nets(smemraw, t1, tid, lane, wave);  // f1, g1 (barriers inside)

    // p1 = f1*dt + g1.dx ; y1 = y + 0.5*(p0 + p1)
#pragma unroll
    for (int rep = 0; rep < 2; ++rep) {
      const int j = tid + rep * THREADS;
      const int m = j >> 6, h = j & 63;
      float a = (float)fbuf[j] * dt;
      const _Float16* gr = gbuf + (m * 512 + h * 8);
      const float* dxr = dxs + m * 8;
#pragma unroll
      for (int cc = 0; cc < 8; ++cc) a += (float)gr[cc] * dxr[cc];
      ybuf[j] = ybuf[j] + 0.5f * (p0buf[j] + a);
    }
    __syncthreads();

    if (tid < MT) {  // readout -> out[i+1] (bank-rotated traversal)
      float a = Fp[F_RB];
      for (int h = 0; h < HID; ++h) {
        const int hh = (h + tid * 4) & 63;
        a += ybuf[tid * HID + hh] * Fp[F_RW + hh];
      }
      out[(size_t)(i + 1) * NBATCH + row0 + tid] = a;
    }
    __syncthreads();
  }
}

extern "C" void kernel_launch(void* const* d_in, const int* in_sizes, int n_in,
                              void* d_out, int out_size, void* d_ws, size_t ws_size,
                              hipStream_t stream) {
  (void)n_in; (void)out_size; (void)d_ws; (void)ws_size;
  const float* ts      = (const float*)d_in[0];
  const float* xinit   = (const float*)d_in[1];
  const float* control = (const float*)d_in[2];
  auto P = [&](int i) { return (const float*)d_in[i]; };

  const float *iW1, *ib1, *iW2, *ib2, *iW3, *ib3;
  const float *vW1, *vb1, *vW2, *vb2, *vW3, *vb3;
  const float *cW1, *cb1, *cW2, *cb2, *cW3, *cb3;
  const float *rW, *rb;

  if (in_sizes[3] == 1024) {
    // params flattened in insertion order: initial, vf, cvf, readout
    iW1 = P(3);  ib1 = P(4);  iW2 = P(5);  ib2 = P(6);  iW3 = P(7);  ib3 = P(8);
    vW1 = P(9);  vb1 = P(10); vW2 = P(11); vb2 = P(12); vW3 = P(13); vb3 = P(14);
    cW1 = P(15); cb1 = P(16); cW2 = P(17); cb2 = P(18); cW3 = P(19); cb3 = P(20);
    rW  = P(21); rb  = P(22);
  } else {
    // jax tree order (dict keys sorted): cvf, initial, readout, vf
    cW1 = P(3);  cb1 = P(4);  cW2 = P(5);  cb2 = P(6);  cW3 = P(7);  cb3 = P(8);
    iW1 = P(9);  ib1 = P(10); iW2 = P(11); ib2 = P(12); iW3 = P(13); ib3 = P(14);
    rW  = P(15); rb  = P(16);
    vW1 = P(17); vb1 = P(18); vW2 = P(19); vb2 = P(20); vW3 = P(21); vb3 = P(22);
  }

  (void)hipFuncSetAttribute((const void*)ncde_kernel,
                            hipFuncAttributeMaxDynamicSharedMemorySize,
                            SMEM_BYTES);
  ncde_kernel<<<dim3(NBATCH / MT), dim3(THREADS), SMEM_BYTES, stream>>>(
      ts, xinit, control,
      vW1, vb1, vW2, vb2, vW3, vb3,
      cW1, cb1, cW2, cb2, cW3, cb3,
      iW1, ib1, iW2, ib2, iW3, ib3,
      rW, rb, (float*)d_out);
}
